// DDFConvGuidedFilter_19198503813622
// MI455X (gfx1250) — matrix operand order repaired and checked by reading, not compiled, not run on hardware
//
#include <hip/hip_runtime.h>

// ---------------------------------------------------------------------------
// DDF ConvGuidedFilter cascade for MI455X (gfx1250, wave32).
//
// Pipeline (per batch image, H=W=512, C=3):
//   K1: s = sum_c x, q = sum_c x^2
//   K2 (xR, eps): tiled stage kernel
//        m = box_R(s)/(3*cnt); v = box_R(q)/(3*cnt) - m^2
//        A = v/(v+eps); b = m - A*m
//        a = 3*box_R(A); beta = 3*box_R(b)
//        s' = a*s + 3*beta; q' = a^2 q + 2 a beta s + 3 beta^2
//   K3: out_o = sum_i w1[o,i] * D_i,  D built per pixel from x and (a_i,beta_i)
//       -> fixed 3x9 GEMM per pixel, executed with v_wmma_f32_16x16x32_f16.
// ---------------------------------------------------------------------------

constexpr int Hh = 512, Ww = 512, Bb = 16;
constexpr int HWc = Hh * Ww;                 // 262144 = 1<<18

static __device__ __forceinline__ int imin(int a, int b) { return a < b ? a : b; }
static __device__ __forceinline__ int imax(int a, int b) { return a > b ? a : b; }

// -------------------- gfx1250 async global->LDS staging ---------------------
static __device__ __forceinline__ void async_load_f32(unsigned lds_byte_off,
                                                      const float* gptr) {
  unsigned long long ga = (unsigned long long)(const void*)gptr;
  // vdst = LDS byte address VGPR, vaddr = 64-bit global address, no saddr.
  asm volatile("global_load_async_to_lds_b32 %0, %1, off"
               :: "v"(lds_byte_off), "v"(ga) : "memory");
}
static __device__ __forceinline__ void wait_async() {
  asm volatile("s_wait_asynccnt 0x0" ::: "memory");
}

// ------------------------------- Kernel 1 ----------------------------------
__global__ void sumsq_kernel(const float* __restrict__ x,
                             float* __restrict__ s, float* __restrict__ q) {
  int p = blockIdx.x * blockDim.x + threadIdx.x;        // p in [0, B*HW)
  int b = p >> 18;
  int pix = p & (HWc - 1);
  const float* xb = x + (size_t)b * 3 * HWc + pix;
  float x0 = xb[0], x1 = xb[HWc], x2 = xb[2 * HWc];
  s[p] = x0 + x1 + x2;
  q[p] = x0 * x0 + x1 * x1 + x2 * x2;
}

// ------------------------------- Kernel 2 ----------------------------------
// One workgroup per TxT output tile per batch image. Halo = 2R (two chained
// radius-R boxes). Separable box sums in LDS; input staged asynchronously.
template <int R, int T>
__global__ void cgf_stage(const float* __restrict__ s_in,
                          const float* __restrict__ q_in,
                          float* __restrict__ a_out, float* __restrict__ b_out,
                          float* __restrict__ s_next, float* __restrict__ q_next,
                          float eps) {
  constexpr int E = T + 4 * R;   // extended tile (s,q)
  constexpr int M = T + 2 * R;   // mid tile (A,b)
  extern __shared__ float smem[];
  float* s_t = smem;             // E*E
  float* q_t = s_t + E * E;      // E*E
  float* sh  = q_t + E * E;      // E*M (row-box of s)
  float* qh  = sh + E * M;       // E*M
  float* A_t = qh + E * M;       // M*M
  float* B_t = A_t + M * M;      // M*M
  float* Ah  = B_t + M * M;      // M*T
  float* Bh  = Ah + M * T;       // M*T

  const int tid = threadIdx.x;
  const int gx0 = blockIdx.x * T - 2 * R;
  const int gy0 = blockIdx.y * T - 2 * R;
  const float* sb = s_in + (size_t)blockIdx.z * HWc;
  const float* qb = q_in + (size_t)blockIdx.z * HWc;

  // Phase 1: stage s,q halo tile into LDS (async for in-bounds, zero pad OOB).
  for (int idx = tid; idx < E * E; idx += blockDim.x) {
    int r = idx / E, c = idx % E;
    int gy = gy0 + r, gx = gx0 + c;
    if (gy >= 0 && gy < Hh && gx >= 0 && gx < Ww) {
      int g = gy * Ww + gx;
      async_load_f32((unsigned)(size_t)(&s_t[idx]), sb + g);
      async_load_f32((unsigned)(size_t)(&q_t[idx]), qb + g);
    } else {
      s_t[idx] = 0.f;
      q_t[idx] = 0.f;
    }
  }
  wait_async();
  __syncthreads();

  // Phase 2: horizontal box sums of s,q.
  for (int idx = tid; idx < E * M; idx += blockDim.x) {
    int r = idx / M, c = idx % M;
    float as = 0.f, aq = 0.f;
#pragma unroll
    for (int d = 0; d <= 2 * R; ++d) {
      as += s_t[r * E + c + d];
      aq += q_t[r * E + c + d];
    }
    sh[idx] = as;
    qh[idx] = aq;
  }
  __syncthreads();

  // Phase 3: vertical box + pointwise A,b (zero outside image, like ref pad).
  for (int idx = tid; idx < M * M; idx += blockDim.x) {
    int r = idx / M, c = idx % M;
    float S = 0.f, Q = 0.f;
#pragma unroll
    for (int d = 0; d <= 2 * R; ++d) {
      S += sh[(r + d) * M + c];
      Q += qh[(r + d) * M + c];
    }
    int gy = gy0 + R + r, gx = gx0 + R + c;
    float A = 0.f, Bv = 0.f;
    if (gy >= 0 && gy < Hh && gx >= 0 && gx < Ww) {
      int ch = imin(gy + R, Hh - 1) - imax(gy - R, 0) + 1;   // closed-form cnt
      int cw = imin(gx + R, Ww - 1) - imax(gx - R, 0) + 1;
      float inv = 1.f / (3.f * (float)(ch * cw));
      float m = S * inv;
      float v = Q * inv - m * m;
      A = v / (v + eps);
      Bv = m - A * m;
    }
    A_t[idx] = A;
    B_t[idx] = Bv;
  }
  __syncthreads();

  // Phase 4: horizontal box of A,b.
  for (int idx = tid; idx < M * T; idx += blockDim.x) {
    int r = idx / T, c = idx % T;
    float aA = 0.f, aB = 0.f;
#pragma unroll
    for (int d = 0; d <= 2 * R; ++d) {
      aA += A_t[r * M + c + d];
      aB += B_t[r * M + c + d];
    }
    Ah[idx] = aA;
    Bh[idx] = aB;
  }
  __syncthreads();

  // Phase 5: vertical box -> a,beta; emit next-stage s,q in closed form.
  for (int idx = tid; idx < T * T; idx += blockDim.x) {
    int r = idx / T, c = idx % T;
    float aA = 0.f, aB = 0.f;
#pragma unroll
    for (int d = 0; d <= 2 * R; ++d) {
      aA += Ah[(r + d) * T + c];
      aB += Bh[(r + d) * T + c];
    }
    float a = 3.f * aA, beta = 3.f * aB;
    int gy = blockIdx.y * T + r, gx = blockIdx.x * T + c;
    size_t go = (size_t)blockIdx.z * HWc + gy * Ww + gx;
    a_out[go] = a;
    b_out[go] = beta;
    if (s_next) {
      float sc = s_t[(2 * R + r) * E + (2 * R + c)];
      float qc = q_t[(2 * R + r) * E + (2 * R + c)];
      s_next[go] = a * sc + 3.f * beta;
      q_next[go] = a * a * qc + 2.f * a * beta * sc + 3.f * beta * beta;
    }
  }
}

// ------------------------------- Kernel 3 ----------------------------------
typedef __attribute__((ext_vector_type(16))) _Float16 v16h;
typedef __attribute__((ext_vector_type(8)))  float    v8f;

__global__ void fuse_out_kernel(const float* __restrict__ x,
                                const float* __restrict__ w1,
                                const float* __restrict__ a1, const float* __restrict__ b1,
                                const float* __restrict__ a2, const float* __restrict__ b2,
                                const float* __restrict__ a3, const float* __restrict__ b3,
                                float* __restrict__ out, int pixPerBlock) {
  const int lane = threadIdx.x & 31;
  const int wave = threadIdx.x >> 5;
  const int nWaves = blockDim.x >> 5;
  const int m = lane & 15;
  const bool lo = lane < 16;
  const float msk = lo ? 1.f : 0.f;

  // A-matrix: w1 (3x9) zero-padded into 16x32 f16 WMMA A layout.
  // Lane l: M = l&15; element e holds K = 2*kb+e (contiguous order).
  v16h Aw = {};
  const int kb = lo ? 0 : 8;
#pragma unroll
  for (int e = 0; e < 16; ++e) {
    int j = e >> 1, t = e & 1;
    int k = 2 * kb + 2 * j + t;
    float wv = 0.f;
    if (m < 3 && k < 9) wv = w1[m * 9 + k];
    Aw[e] = (_Float16)wv;
  }

  const long base = (long)blockIdx.x * pixPerBlock;
  for (int it = wave * 32; it < pixPerBlock; it += nWaves * 32) {
    long p = base + it + lane;                 // 32 coalesced pixels per wave
    int b = (int)(p >> 18);
    int pix = (int)(p & (HWc - 1));
    size_t mo = (size_t)b * HWc + pix;
    size_t xo = (size_t)b * 3 * HWc + pix;

    float x0 = x[xo], x1 = x[xo + HWc], x2 = x[xo + 2 * HWc];
    float A1 = a1[mo], B1 = b1[mo];
    float A2 = a2[mo], B2 = b2[mo];
    float A3 = a3[mo], B3 = b3[mo];

    float F10 = A1 * x0 + B1, F11 = A1 * x1 + B1, F12 = A1 * x2 + B1;
    float F20 = A2 * F10 + B2, F21 = A2 * F11 + B2, F22 = A2 * F12 + B2;
    float F30 = A3 * F20 + B3, F31 = A3 * F21 + B3, F32 = A3 * F22 + B3;
    float D[9] = { x0 - F10, x1 - F11, x2 - F12,
                   F10 - F20, F11 - F21, F12 - F22,
                   F20 - F30, F21 - F31, F22 - F32 };

    // WMMA #1: pixels of lanes 0..15 (B cols = lanes 0..15, K rows 0..8).
    v16h Bv = {};
#pragma unroll
    for (int e = 0; e < 9; ++e) Bv[e] = (_Float16)(D[e] * msk);
    v8f c0 = {};
    c0 = __builtin_amdgcn_wmma_f32_16x16x32_f16(false, Aw, false, Bv,
                                                (short)0, c0, false, false);

    // WMMA #2: pixels of lanes 16..31, swapped into low-lane B positions.
    v16h Bw = {};
#pragma unroll
    for (int e = 0; e < 9; ++e) {
      float ds = __shfl_xor(D[e], 16, 32);
      Bw[e] = (_Float16)(ds * msk);
    }
    v8f c1 = {};
    c1 = __builtin_amdgcn_wmma_f32_16x16x32_f16(false, Aw, false, Bw,
                                                (short)0, c1, false, false);

    if (lo) {
      long p0 = base + it + m;
      int b0 = (int)(p0 >> 18);
      int q0 = (int)(p0 & (HWc - 1));
      size_t o0 = (size_t)b0 * 3 * HWc + q0;
      out[o0] = c0[0]; out[o0 + HWc] = c0[1]; out[o0 + 2 * HWc] = c0[2];

      long p1 = base + it + 16 + m;
      int b1i = (int)(p1 >> 18);
      int q1 = (int)(p1 & (HWc - 1));
      size_t o1 = (size_t)b1i * 3 * HWc + q1;
      out[o1] = c1[0]; out[o1 + HWc] = c1[1]; out[o1 + 2 * HWc] = c1[2];
    }
  }
}

// ------------------------------- launcher ----------------------------------
static inline size_t stage_lds_bytes(int R, int T) {
  int E = T + 4 * R, M = T + 2 * R;
  return sizeof(float) * (size_t)(2 * E * E + 2 * E * M + 2 * M * M + 2 * M * T);
}

extern "C" void kernel_launch(void* const* d_in, const int* in_sizes, int n_in,
                              void* d_out, int out_size, void* d_ws, size_t ws_size,
                              hipStream_t stream) {
  (void)in_sizes; (void)n_in; (void)out_size; (void)ws_size;
  const float* x  = (const float*)d_in[0];
  const float* w1 = (const float*)d_in[1];
  float* out = (float*)d_out;
  float* ws = (float*)d_ws;

  const size_t MAP = (size_t)Bb * HWc;   // 4,194,304 elems per map
  float* sA = ws + 0 * MAP;  float* qA = ws + 1 * MAP;
  float* sB = ws + 2 * MAP;  float* qB = ws + 3 * MAP;
  float* a1 = ws + 4 * MAP;  float* b1 = ws + 5 * MAP;
  float* a2 = ws + 6 * MAP;  float* b2 = ws + 7 * MAP;
  float* a3 = ws + 8 * MAP;  float* b3 = ws + 9 * MAP;

  sumsq_kernel<<<(int)(MAP / 256), 256, 0, stream>>>(x, sA, qA);

  dim3 g32(Ww / 32, Hh / 32, Bb);
  dim3 g16(Ww / 16, Hh / 16, Bb);
  cgf_stage<1, 32><<<g32, 256, stage_lds_bytes(1, 32), stream>>>(
      sA, qA, a1, b1, sB, qB, 0.16f);
  cgf_stage<3, 32><<<g32, 256, stage_lds_bytes(3, 32), stream>>>(
      sB, qB, a2, b2, sA, qA, 0.04f);
  cgf_stage<7, 16><<<g16, 256, stage_lds_bytes(7, 16), stream>>>(
      sA, qA, a3, b3, nullptr, nullptr, 0.01f);

  fuse_out_kernel<<<1024, 256, 0, stream>>>(x, w1, a1, b1, a2, b2, a3, b3,
                                            out, 4096);
}
